// MoELayer_49478023250021
// MI455X (gfx1250) — compile-verified, matrix operand order checked by reading, not run on hardware
//
#include <hip/hip_runtime.h>
#include <stdint.h>

#define D_DIM 1024
#define H_DIM 2048
#define E_NUM 8
#define N_TOK 4096
#define TM 128
#define CAP (N_TOK * 2 + E_NUM * TM)   /* 9216 routed-row capacity */
#define MAX_TILES (CAP / TM)           /* 72 */
#define BSTRIDE 40                     /* LDS row stride in bf16 elems (pad vs banks) */

typedef __attribute__((ext_vector_type(16))) __bf16 v16bf;
typedef __attribute__((ext_vector_type(8)))  float  v8f;

union FragBF { v16bf v; uint4 q[2]; };

__device__ __forceinline__ unsigned bf16pk(float a, float b) {
#if defined(__has_builtin) && __has_builtin(__builtin_amdgcn_cvt_pk_bf16_f32)
  typedef __attribute__((ext_vector_type(2))) __bf16 v2bf_t;
  union { v2bf_t v; unsigned u; } cv;
  cv.v = __builtin_amdgcn_cvt_pk_bf16_f32(a, b);
  return cv.u;
#else
  unsigned ua = __float_as_uint(a), ub = __float_as_uint(b);
  ua = ua + 0x7FFFu + ((ua >> 16) & 1u);
  ub = ub + 0x7FFFu + ((ub >> 16) & 1u);
  return __builtin_amdgcn_perm(ub, ua, 0x07060302u);  // {hi16(ub),hi16(ua)}
#endif
}

__device__ __forceinline__ uint16_t bf16one(float a) {
  unsigned u = __float_as_uint(a);
  u = (u + 0x7FFFu + ((u >> 16) & 1u)) >> 16;
  return (uint16_t)u;
}

// ---------------- init ----------------
__global__ void moe_init(int* row_token, int* counts, int* cursors) {
  int i = blockIdx.x * 256 + threadIdx.x;
  if (i < CAP) row_token[i] = -1;
  if (i < E_NUM) { counts[i] = 0; cursors[i] = 0; }
}

// ---------------- x -> bf16 ----------------
__global__ void moe_xcvt(const float* __restrict__ x, uint16_t* __restrict__ xbf) {
  size_t i = ((size_t)blockIdx.x * blockDim.x + threadIdx.x) * 4;
  float4 f = *(const float4*)(x + i);
  uint2 o; o.x = bf16pk(f.x, f.y); o.y = bf16pk(f.z, f.w);
  *(uint2*)(xbf + i) = o;
}

// ---------------- gating: one wave per token ----------------
__global__ __launch_bounds__(256) void moe_gate(const float* __restrict__ x,
                                                const float* __restrict__ gw,
                                                int* __restrict__ counts,
                                                int* __restrict__ tok_e,
                                                float* __restrict__ tok_g) {
  int n    = blockIdx.x * 8 + (threadIdx.x >> 5);
  int lane = threadIdx.x & 31;
  const float* xr = x + (size_t)n * D_DIM;
  float acc[E_NUM];
#pragma unroll
  for (int e = 0; e < E_NUM; ++e) acc[e] = 0.f;
  for (int i = 0; i < D_DIM / 32; ++i) {
    float xv = xr[lane + 32 * i];
#pragma unroll
    for (int e = 0; e < E_NUM; ++e)
      acc[e] = fmaf(xv, gw[e * D_DIM + lane + 32 * i], acc[e]);
  }
#pragma unroll
  for (int off = 16; off >= 1; off >>= 1)
#pragma unroll
    for (int e = 0; e < E_NUM; ++e) acc[e] += __shfl_xor(acc[e], off, 32);
  if (lane == 0) {
    int b0 = -1, b1 = -1; float v0 = -3.4e38f, v1 = -3.4e38f;
#pragma unroll
    for (int e = 0; e < E_NUM; ++e) {
      float v = acc[e];
      if (v > v0)      { v1 = v0; b1 = b0; v0 = v; b0 = e; }
      else if (v > v1) { v1 = v; b1 = e; }
    }
    float g0 = 1.f / (1.f + __expf(v1 - v0));   // softmax over top-2
    float g1 = 1.f - g0;
    atomicAdd(&counts[b0], 1); atomicAdd(&counts[b1], 1);
    tok_e[2 * n] = b0; tok_e[2 * n + 1] = b1;
    tok_g[2 * n] = g0; tok_g[2 * n + 1] = g1;
  }
}

// ---------------- prefix: aligned offsets + tile->expert ----------------
__global__ void moe_prefix(const int* __restrict__ counts, int* __restrict__ offs,
                           int* __restrict__ tile_expert, int* __restrict__ cursors) {
  __shared__ int so[E_NUM + 1];
  if (threadIdx.x == 0) {
    int o = 0;
    for (int e = 0; e < E_NUM; ++e) {
      so[e] = o; offs[e] = o;
      o += (counts[e] + TM - 1) & ~(TM - 1);
    }
    so[E_NUM] = o; offs[E_NUM] = o;
  }
  __syncthreads();
  for (int t = threadIdx.x; t < MAX_TILES; t += blockDim.x) {
    int base = t * TM, e = -1;
    for (int q = 0; q < E_NUM; ++q)
      if (base >= so[q] && base < so[q + 1]) e = q;
    tile_expert[t] = e;
  }
  if (threadIdx.x < E_NUM) cursors[threadIdx.x] = 0;
}

// ---------------- scatter tokens into routed rows ----------------
__global__ void moe_scatter(const int* __restrict__ tok_e, const float* __restrict__ tok_g,
                            const int* __restrict__ offs, int* __restrict__ cursors,
                            int* __restrict__ row_token, float* __restrict__ row_gate,
                            int* __restrict__ token_slot) {
  int n = blockIdx.x * blockDim.x + threadIdx.x;
  if (n >= N_TOK) return;
#pragma unroll
  for (int kk = 0; kk < 2; ++kk) {
    int e   = tok_e[2 * n + kk];
    int pos = offs[e] + atomicAdd(&cursors[e], 1);
    row_token[pos]  = n;
    row_gate[pos]   = tok_g[2 * n + kk];
    token_slot[2 * n + kk] = pos;
  }
}

// ---- shared helper: convert 16 f32 -> 16 bf16 and store to LDS ----
__device__ __forceinline__ void stage_store(uint16_t* dst, const float4& f0,
                                            const float4& f1, const float4& f2,
                                            const float4& f3) {
  uint4 o0, o1;
  o0.x = bf16pk(f0.x, f0.y); o0.y = bf16pk(f0.z, f0.w);
  o0.z = bf16pk(f1.x, f1.y); o0.w = bf16pk(f1.z, f1.w);
  o1.x = bf16pk(f2.x, f2.y); o1.y = bf16pk(f2.z, f2.w);
  o1.z = bf16pk(f3.x, f3.y); o1.w = bf16pk(f3.z, f3.w);
  ((uint4*)dst)[0] = o0; ((uint4*)dst)[1] = o1;
}

// ---------------- GEMM1: h = silu(x @ w1^T + b1), bf16 out ----------------
// Padding rows (row_token == -1) are clamped to token 0: WMMA output row m depends
// only on A row m, so padding rows produce values that are simply never combined.
__global__ __launch_bounds__(256) void moe_gemm1(
    const uint16_t* __restrict__ xbf, const float* __restrict__ w1,
    const float* __restrict__ b1, uint16_t* __restrict__ hbuf,
    const int* __restrict__ row_token, const int* __restrict__ tile_expert) {
  const int tmb = blockIdx.x;
  const int e = tile_expert[tmb];
  if (e < 0) return;
  const int tid = threadIdx.x, wave = tid >> 5, lane = tid & 31;
  const int m = lane & 15, khalf = lane >> 4, k0 = khalf * 8;
  const int rowbase = tmb * TM + wave * 16;
  int tok = row_token[rowbase + m];
  tok = tok < 0 ? 0 : tok;
  const uint16_t* arow = xbf + (size_t)tok * D_DIM + k0;

  __shared__ uint16_t bsh[2][128 * BSTRIDE];
  const int fn = tid >> 1, fk = (tid & 1) * 16;
  const float* bsrc = w1 + (size_t)e * H_DIM * D_DIM
                         + (size_t)(blockIdx.y * 128 + fn) * D_DIM + fk;
  uint16_t* fdst0 = &bsh[0][fn * BSTRIDE + fk];
  uint16_t* fdst1 = &bsh[1][fn * BSTRIDE + fk];

  v8f acc[8];
#pragma unroll
  for (int t = 0; t < 8; ++t)
#pragma unroll
    for (int j = 0; j < 8; ++j) acc[t][j] = 0.f;

  uint4 a0 = *(const uint4*)(arow);       // A(k=0)
  uint4 a1 = *(const uint4*)(arow + 16);
  {   // prologue: stage k=0 into buffer 0
    const float4* p = (const float4*)(bsrc);
    stage_store(fdst0, p[0], p[1], p[2], p[3]);
  }
  __syncthreads();

  int kb = 0;
  for (int k = 0; k < D_DIM; k += 32, kb ^= 1) {
    const bool have_next = (k + 32) < D_DIM;
    float4 nf0, nf1, nf2, nf3;
    uint4 na0, na1;
    if (have_next) {           // global prefetch of next A + B chunks
      const float4* p = (const float4*)(bsrc + k + 32);
      nf0 = p[0]; nf1 = p[1]; nf2 = p[2]; nf3 = p[3];
      na0 = *(const uint4*)(arow + k + 32);
      na1 = *(const uint4*)(arow + k + 48);
    }
    FragBF a; a.q[0] = a0; a.q[1] = a1;
    const uint16_t* bb = bsh[kb];
#pragma unroll
    for (int t = 0; t < 8; ++t) {
      FragBF b;
      b.q[0] = *(const uint4*)&bb[(t * 16 + m) * BSTRIDE + k0];
      b.q[1] = *(const uint4*)&bb[(t * 16 + m) * BSTRIDE + k0 + 16];
      acc[t] = __builtin_amdgcn_wmma_f32_16x16x32_bf16(
          false, a.v, false, b.v, (short)0, acc[t], false, false);
    }
    if (have_next) {
      stage_store(kb ? fdst0 : fdst1, nf0, nf1, nf2, nf3);
      a0 = na0; a1 = na1;
    }
    __syncthreads();
  }

  const int colb = blockIdx.y * 128;
#pragma unroll
  for (int t = 0; t < 8; ++t) {
    const int col = colb + t * 16 + m;
    const float bias = b1[e * H_DIM + col];
#pragma unroll
    for (int j = 0; j < 8; ++j) {
      const int mr = j + khalf * 8;
      float v = acc[t][j] + bias;
      float s = v / (1.f + __expf(-v));
      hbuf[(size_t)(rowbase + mr) * H_DIM + col] = bf16one(s);
    }
  }
}

// ---------------- GEMM2: y = gate * (h @ w2^T + b2), f32 out ----------------
__global__ __launch_bounds__(256) void moe_gemm2(
    const uint16_t* __restrict__ hbuf, const float* __restrict__ w2,
    const float* __restrict__ b2, float* __restrict__ ybuf,
    const float* __restrict__ row_gate, const int* __restrict__ tile_expert) {
  const int tmb = blockIdx.x;
  const int e = tile_expert[tmb];
  if (e < 0) return;
  const int tid = threadIdx.x, wave = tid >> 5, lane = tid & 31;
  const int m = lane & 15, khalf = lane >> 4, k0 = khalf * 8;
  const int rowbase = tmb * TM + wave * 16;
  const uint16_t* arow = hbuf + (size_t)(rowbase + m) * H_DIM + k0;

  __shared__ uint16_t bsh[2][128 * BSTRIDE];
  const int fn = tid >> 1, fk = (tid & 1) * 16;
  const float* bsrc = w2 + (size_t)e * D_DIM * H_DIM
                         + (size_t)(blockIdx.y * 128 + fn) * H_DIM + fk;
  uint16_t* fdst0 = &bsh[0][fn * BSTRIDE + fk];
  uint16_t* fdst1 = &bsh[1][fn * BSTRIDE + fk];

  v8f acc[8];
#pragma unroll
  for (int t = 0; t < 8; ++t)
#pragma unroll
    for (int j = 0; j < 8; ++j) acc[t][j] = 0.f;

  uint4 a0 = *(const uint4*)(arow);
  uint4 a1 = *(const uint4*)(arow + 16);
  {
    const float4* p = (const float4*)(bsrc);
    stage_store(fdst0, p[0], p[1], p[2], p[3]);
  }
  __syncthreads();

  int kb = 0;
  for (int k = 0; k < H_DIM; k += 32, kb ^= 1) {
    const bool have_next = (k + 32) < H_DIM;
    float4 nf0, nf1, nf2, nf3;
    uint4 na0, na1;
    if (have_next) {
      const float4* p = (const float4*)(bsrc + k + 32);
      nf0 = p[0]; nf1 = p[1]; nf2 = p[2]; nf3 = p[3];
      na0 = *(const uint4*)(arow + k + 32);
      na1 = *(const uint4*)(arow + k + 48);
    }
    FragBF a; a.q[0] = a0; a.q[1] = a1;
    const uint16_t* bb = bsh[kb];
#pragma unroll
    for (int t = 0; t < 8; ++t) {
      FragBF b;
      b.q[0] = *(const uint4*)&bb[(t * 16 + m) * BSTRIDE + k0];
      b.q[1] = *(const uint4*)&bb[(t * 16 + m) * BSTRIDE + k0 + 16];
      acc[t] = __builtin_amdgcn_wmma_f32_16x16x32_bf16(
          false, a.v, false, b.v, (short)0, acc[t], false, false);
    }
    if (have_next) {
      stage_store(kb ? fdst0 : fdst1, nf0, nf1, nf2, nf3);
      a0 = na0; a1 = na1;
    }
    __syncthreads();
  }

  float g[8];
#pragma unroll
  for (int j = 0; j < 8; ++j) g[j] = row_gate[rowbase + j + khalf * 8];

  const int colb = blockIdx.y * 128;
#pragma unroll
  for (int t = 0; t < 8; ++t) {
    const int col = colb + t * 16 + m;
    const float bias = b2[e * D_DIM + col];
#pragma unroll
    for (int j = 0; j < 8; ++j) {
      const int mr = j + khalf * 8;
      ybuf[(size_t)(rowbase + mr) * D_DIM + col] = g[j] * (acc[t][j] + bias);
    }
  }
}

// ---------------- combine the 2 routed rows per token ----------------
__global__ void moe_combine(const float* __restrict__ ybuf,
                            const int* __restrict__ token_slot,
                            float* __restrict__ out) {
  int n = blockIdx.x;
  int d = threadIdx.x * 4;
  int s0 = token_slot[2 * n], s1 = token_slot[2 * n + 1];
  float4 a = *(const float4*)(ybuf + (size_t)s0 * D_DIM + d);
  float4 b = *(const float4*)(ybuf + (size_t)s1 * D_DIM + d);
  float4 o; o.x = a.x + b.x; o.y = a.y + b.y; o.z = a.z + b.z; o.w = a.w + b.w;
  *(float4*)(out + (size_t)n * D_DIM + d) = o;
}

// ---------------- aux balance loss ----------------
__global__ void moe_aux(const int* __restrict__ counts, float* __restrict__ out_aux) {
  if (threadIdx.x == 0) {
    float s = 0.f;
    for (int e = 0; e < E_NUM; ++e) {
      float l = (float)counts[e] / (float)(N_TOK * 2);
      s += l * l;
    }
    out_aux[0] = 0.01f * (float)E_NUM * s;
  }
}

extern "C" void kernel_launch(void* const* d_in, const int* in_sizes, int n_in,
                              void* d_out, int out_size, void* d_ws, size_t ws_size,
                              hipStream_t stream) {
  const float* x  = (const float*)d_in[0];
  const float* gw = (const float*)d_in[1];
  const float* w1 = (const float*)d_in[2];
  const float* b1 = (const float*)d_in[3];
  const float* w2 = (const float*)d_in[4];
  const float* b2 = (const float*)d_in[5];
  float* out = (float*)d_out;

  // ---- workspace layout (~84 MB) ----
  char* ws = (char*)d_ws;
  size_t cur = 0;
  auto walloc = [&](size_t bytes) -> void* {
    void* p = ws + cur;
    cur = (cur + bytes + 255) & ~(size_t)255;
    return p;
  };
  uint16_t* xbf       = (uint16_t*)walloc((size_t)N_TOK * D_DIM * 2);
  uint16_t* hbuf      = (uint16_t*)walloc((size_t)CAP * H_DIM * 2);
  float*    ybuf      = (float*)   walloc((size_t)CAP * D_DIM * 4);
  int*      row_token = (int*)     walloc((size_t)CAP * 4);
  float*    row_gate  = (float*)   walloc((size_t)CAP * 4);
  int*      token_slot= (int*)     walloc((size_t)N_TOK * 2 * 4);
  int*      tok_e     = (int*)     walloc((size_t)N_TOK * 2 * 4);
  float*    tok_g     = (float*)   walloc((size_t)N_TOK * 2 * 4);
  int*      counts    = (int*)     walloc(E_NUM * 4);
  int*      cursors   = (int*)     walloc(E_NUM * 4);
  int*      offs      = (int*)     walloc((E_NUM + 1) * 4);
  int*      tile_exp  = (int*)     walloc(MAX_TILES * 4);

  moe_init<<<(CAP + 255) / 256, 256, 0, stream>>>(row_token, counts, cursors);
  moe_xcvt<<<((size_t)N_TOK * D_DIM / 4 + 255) / 256, 256, 0, stream>>>(x, xbf);
  moe_gate<<<N_TOK / 8, 256, 0, stream>>>(x, gw, counts, tok_e, tok_g);
  moe_prefix<<<1, 256, 0, stream>>>(counts, offs, tile_exp, cursors);
  moe_scatter<<<(N_TOK + 255) / 256, 256, 0, stream>>>(tok_e, tok_g, offs, cursors,
                                                       row_token, row_gate, token_slot);
  dim3 g1(MAX_TILES, H_DIM / 128);
  moe_gemm1<<<g1, 256, 0, stream>>>(xbf, w1, b1, hbuf, row_token, tile_exp);
  dim3 g2(MAX_TILES, D_DIM / 128);
  moe_gemm2<<<g2, 256, 0, stream>>>(hbuf, w2, b2, ybuf, row_gate, tile_exp);
  moe_combine<<<N_TOK, 256, 0, stream>>>(ybuf, token_slot, out);
  moe_aux<<<1, 32, 0, stream>>>(counts, out + (size_t)N_TOK * D_DIM);
}
